// PointNetFeaturePropagation_14766097564244
// MI455X (gfx1250) — matrix operand order repaired and checked by reading, not compiled, not run on hardware
//
#include <hip/hip_runtime.h>
#include <hip/hip_bf16.h>
#include <stdint.h>

// ---------------------------------------------------------------------------
// MI455X (gfx1250, wave32) implementation of PointNetFeaturePropagation.
// Distance/top-3 pass: VALU kernel with xyz2 staged in LDS (broadcast reads).
// Conv1x1+BN+ReLU chain: bf16 WMMA GEMMs (v_wmma_f32_16x16x32_bf16) with
// folded BN scale/shift epilogue, fused residual, bf16 activations between
// layers (all resident in the 192MB L2).
// ---------------------------------------------------------------------------

typedef __attribute__((ext_vector_type(16))) __bf16 v16bf;
typedef __attribute__((ext_vector_type(8)))  float  v8f;

#define BB   2
#define NN   16384
#define SS   4096
#define D1C  128
#define D2C  256
#define CIN  397          // D1 + D2 + 13
#define KP0  416          // CIN padded to multiple of 32
#define CC   128

__device__ __forceinline__ unsigned short f2bf(float f) {
  unsigned int u = __float_as_uint(f);
  unsigned int r = (u + 0x7FFFu + ((u >> 16) & 1u)) >> 16;   // RNE
  return (unsigned short)r;
}
__device__ __forceinline__ float bf2f(unsigned short h) {
  return __uint_as_float(((unsigned int)h) << 16);
}

// ---- prep: pad/convert weights to bf16 [coutp][kp] --------------------------
__global__ void prep_weight(const float* __restrict__ W, int cout, int cin,
                            unsigned short* __restrict__ Wb, int coutp, int kp) {
  int i = blockIdx.x * blockDim.x + threadIdx.x;
  int tot = coutp * kp;
  if (i >= tot) return;
  int co = i / kp, k = i - co * kp;
  float v = (co < cout && k < cin) ? W[co * cin + k] : 0.f;
  Wb[i] = f2bf(v);
}

// ---- prep: fold conv bias + inference BN into per-channel scale/shift -------
__global__ void prep_affine(const float* __restrict__ bias, const float* __restrict__ bn,
                            int c, float* __restrict__ scale, float* __restrict__ shift,
                            int cp) {
  int i = blockIdx.x * blockDim.x + threadIdx.x;
  if (i >= cp) return;
  if (i < c) {
    float g = bn[0 * c + i], be = bn[1 * c + i], m = bn[2 * c + i], v = bn[3 * c + i];
    float s = g * rsqrtf(v + 1e-5f);
    scale[i] = s;
    shift[i] = be + (bias[i] - m) * s;
  } else {
    scale[i] = 0.f; shift[i] = 0.f;
  }
}

// ---- 3-NN inverse-distance interpolation; builds X0 [B*N][KP0] bf16 ---------
__global__ __launch_bounds__(256)
void interp_kernel(const float* __restrict__ xyz1, const float* __restrict__ xyz2,
                   const float* __restrict__ points1, const float* __restrict__ points2,
                   const float* __restrict__ last_pred, unsigned short* __restrict__ X0) {
  __shared__ float sx[SS * 3];                       // 48 KB, broadcast reads
  const int b = blockIdx.y;
  for (int i = threadIdx.x; i < SS * 3; i += 256) sx[i] = xyz2[(size_t)b * SS * 3 + i];
  __syncthreads();

  const int n = blockIdx.x * 256 + threadIdx.x;
  const float px = xyz1[((size_t)b * NN + n) * 3 + 0];
  const float py = xyz1[((size_t)b * NN + n) * 3 + 1];
  const float pz = xyz1[((size_t)b * NN + n) * 3 + 2];

  float d0 = 3.4e38f, d1 = 3.4e38f, d2 = 3.4e38f;
  int   i0 = 0, i1 = 0, i2 = 0;
  for (int s = 0; s < SS; ++s) {
    float dx = px - sx[s * 3 + 0];
    float dy = py - sx[s * 3 + 1];
    float dz = pz - sx[s * 3 + 2];
    float d = dx * dx + dy * dy + dz * dz;
    if (d < d2) {
      if (d < d1) {
        if (d < d0) { d2 = d1; i2 = i1; d1 = d0; i1 = i0; d0 = d; i0 = s; }
        else        { d2 = d1; i2 = i1; d1 = d;  i1 = s; }
      } else        { d2 = d;  i2 = s; }
    }
  }
  float r0 = 1.f / (d0 + 1e-8f), r1 = 1.f / (d1 + 1e-8f), r2 = 1.f / (d2 + 1e-8f);
  float rs = 1.f / (r0 + r1 + r2);
  float w0 = r0 * rs, w1 = r1 * rs, w2 = r2 * rs;

  unsigned short* row = X0 + ((size_t)b * NN + n) * KP0;
  const float* p1 = points1 + (size_t)b * D1C * NN;
#pragma unroll 4
  for (int c = 0; c < D1C; ++c) row[c] = f2bf(p1[(size_t)c * NN + n]);

  const float* p2 = points2 + (size_t)b * D2C * SS;  // gathers served by L2
#pragma unroll 4
  for (int c = 0; c < D2C; ++c) {
    const float* pc = p2 + (size_t)c * SS;
    row[D1C + c] = f2bf(w0 * pc[i0] + w1 * pc[i1] + w2 * pc[i2]);
  }
  const float* lp = last_pred + (size_t)b * SS * 13;
  for (int j = 0; j < 13; ++j)
    row[D1C + D2C + j] = f2bf(w0 * lp[i0 * 13 + j] + w1 * lp[i1 * 13 + j] + w2 * lp[i2 * 13 + j]);
  for (int c = CIN; c < KP0; ++c) row[c] = 0;
}

// ---- WMMA bf16 GEMM + BN/ReLU (+residual) -----------------------------------
// out[co,n] = relu(scale[co]*sum_k W[co,k]*X[n,k] + shift[co] (+ res[n,co]))
// Per wave: A-frag (16 rows of W) reused across a 16x64 strip (4 wmma/kstep).
union FragU { uint4 q[2]; v16bf v; };

template <int KP, int MB, int NG, bool RES, int OUTMODE, bool WBF>
__global__ __launch_bounds__(MB * NG * 32)
void gemm_bn_relu(const unsigned short* __restrict__ Xin,   // [B*N][KP] bf16
                  const unsigned short* __restrict__ Wb,    // [MB*16][KP] bf16
                  const float* __restrict__ scale, const float* __restrict__ shift,
                  const unsigned short* __restrict__ Xres,  // [B*N][128] bf16 or null
                  unsigned short* __restrict__ Xout,        // [B*N][MB*16] bf16 or null
                  float* __restrict__ outF,                 // f32 out or null
                  int coutReal) {
  static_assert(KP % 32 == 0, "KP must be multiple of 32");
  constexpr int THREADS = MB * NG * 32;
  constexpr int NSPAN = NG * 64;
  constexpr int NT = 4;
  constexpr int COUTP = MB * 16;

  __shared__ alignas(16) unsigned short xs[NSPAN * KP];

  const int b = blockIdx.y;
  const int n_block = blockIdx.x * NSPAN;
  const int tid = threadIdx.x;

  // stage activation tile: rows contiguous -> fully coalesced b128 copies
  {
    const uint4* __restrict__ src = (const uint4*)(Xin + ((size_t)b * NN + n_block) * KP);
    uint4* dst = (uint4*)xs;
    constexpr int CNT = NSPAN * KP / 8;
    for (int i = tid; i < CNT; i += THREADS) dst[i] = src[i];
  }
  __syncthreads();

  const int wave = tid >> 5;
  const int lane = tid & 31;
  const int mb = wave % MB;          // output-channel block
  const int ng = wave / MB;          // n-strip group
  const int lan16 = lane & 15;
  const int khi = lane >> 4;
  const int m0 = mb * 16;

  v8f zero = {0.f, 0.f, 0.f, 0.f, 0.f, 0.f, 0.f, 0.f};
  v8f acc[NT];
#pragma unroll
  for (int t = 0; t < NT; ++t) acc[t] = zero;

  // A fragment source: row m = m0+lan16; 16 contiguous K per lane-half
  const unsigned short* __restrict__ wrow = Wb + (size_t)(m0 + lan16) * KP;

  for (int k0 = 0; k0 < KP; k0 += 32) {
    FragU a;
    a.q[0] = *(const uint4*)(wrow + k0 + khi * 16);
    a.q[1] = *(const uint4*)(wrow + k0 + khi * 16 + 8);
#pragma unroll
    for (int t = 0; t < NT; ++t) {
      // B fragment: column n = lan16, 16 contiguous K per lane-half, from LDS
      const unsigned short* xrow =
          xs + (size_t)(ng * 64 + t * 16 + lan16) * KP + k0 + khi * 16;
      FragU bf;
      bf.q[0] = *(const uint4*)(xrow);
      bf.q[1] = *(const uint4*)(xrow + 8);
      acc[t] = __builtin_amdgcn_wmma_f32_16x16x32_bf16(
          false, a.v, false, bf.v, (short)0, acc[t], false, false);
    }
  }

  // epilogue: C/D layout -> co = m0 + r + 8*(lane>=16), n = lane&15
#pragma unroll
  for (int t = 0; t < NT; ++t) {
#pragma unroll
    for (int r = 0; r < 8; ++r) {
      const int co = m0 + r + khi * 8;
      const int n = n_block + ng * 64 + t * 16 + lan16;
      float v = acc[t][r] * scale[co] + shift[co];
      if (RES) v += bf2f(Xres[((size_t)b * NN + n) * CC + co]);
      v = fmaxf(v, 0.f);
      if (WBF) Xout[((size_t)b * NN + n) * COUTP + co] = f2bf(v);
      if (OUTMODE == 1) outF[((size_t)b * CC + co) * NN + n] = v;              // [B,C,N]
      if (OUTMODE == 2) {
        if (co < coutReal) outF[((size_t)b * NN + n) * coutReal + co] = v;     // [B,N,C]
      }
    }
  }
}

// ---------------------------------------------------------------------------
extern "C" void kernel_launch(void* const* d_in, const int* in_sizes, int n_in,
                              void* d_out, int out_size, void* d_ws, size_t ws_size,
                              hipStream_t stream) {
  (void)in_sizes; (void)n_in; (void)out_size; (void)ws_size;
  const float* xyz1      = (const float*)d_in[0];
  const float* xyz2      = (const float*)d_in[1];
  const float* points1   = (const float*)d_in[2];
  const float* points2   = (const float*)d_in[3];
  const float* last_pred = (const float*)d_in[4];
  const float* fuse_W  = (const float*)d_in[5];
  const float* fuse_b  = (const float*)d_in[6];
  const float* fuse_bn = (const float*)d_in[7];
  const float* ext1_W  = (const float*)d_in[8];
  const float* ext1_b  = (const float*)d_in[9];
  const float* ext1_bn = (const float*)d_in[10];
  const float* ext2_W  = (const float*)d_in[11];
  const float* ext2_b  = (const float*)d_in[12];
  const float* ext2_bn = (const float*)d_in[13];
  const float* pred1_W  = (const float*)d_in[14];
  const float* pred1_b  = (const float*)d_in[15];
  const float* pred1_bn = (const float*)d_in[16];
  const float* pred2_W  = (const float*)d_in[17];
  const float* pred2_b  = (const float*)d_in[18];
  const float* pred2_bn = (const float*)d_in[19];

  char* ws = (char*)d_ws;
  size_t off = 0;
  auto alloc = [&](size_t bytes) -> void* {
    void* p = ws + off;
    off = (off + bytes + 255) & ~(size_t)255;
    return p;
  };

  unsigned short* X0 = (unsigned short*)alloc((size_t)BB * NN * KP0 * 2);
  unsigned short* X1 = (unsigned short*)alloc((size_t)BB * NN * CC * 2);
  unsigned short* H  = (unsigned short*)alloc((size_t)BB * NN * CC * 2);
  unsigned short* XR = (unsigned short*)alloc((size_t)BB * NN * CC * 2);
  unsigned short* F  = (unsigned short*)alloc((size_t)BB * NN * CC * 2);
  unsigned short* Wf  = (unsigned short*)alloc((size_t)CC * KP0 * 2);
  unsigned short* We1 = (unsigned short*)alloc((size_t)CC * CC * 2);
  unsigned short* We2 = (unsigned short*)alloc((size_t)CC * CC * 2);
  unsigned short* Wp1 = (unsigned short*)alloc((size_t)CC * CC * 2);
  unsigned short* Wp2 = (unsigned short*)alloc((size_t)16 * CC * 2);
  float* sc_f  = (float*)alloc(CC * 4);  float* sh_f  = (float*)alloc(CC * 4);
  float* sc_e1 = (float*)alloc(CC * 4);  float* sh_e1 = (float*)alloc(CC * 4);
  float* sc_e2 = (float*)alloc(CC * 4);  float* sh_e2 = (float*)alloc(CC * 4);
  float* sc_p1 = (float*)alloc(CC * 4);  float* sh_p1 = (float*)alloc(CC * 4);
  float* sc_p2 = (float*)alloc(16 * 4);  float* sh_p2 = (float*)alloc(16 * 4);

  // ---- weight / affine prep (tiny) ----
  prep_weight<<<dim3((CC * KP0 + 255) / 256), 256, 0, stream>>>(fuse_W, CC, CIN, Wf, CC, KP0);
  prep_weight<<<dim3((CC * CC + 255) / 256), 256, 0, stream>>>(ext1_W, CC, CC, We1, CC, CC);
  prep_weight<<<dim3((CC * CC + 255) / 256), 256, 0, stream>>>(ext2_W, CC, CC, We2, CC, CC);
  prep_weight<<<dim3((CC * CC + 255) / 256), 256, 0, stream>>>(pred1_W, CC, CC, Wp1, CC, CC);
  prep_weight<<<dim3((16 * CC + 255) / 256), 256, 0, stream>>>(pred2_W, 13, CC, Wp2, 16, CC);
  prep_affine<<<1, 128, 0, stream>>>(fuse_b, fuse_bn, CC, sc_f, sh_f, CC);
  prep_affine<<<1, 128, 0, stream>>>(ext1_b, ext1_bn, CC, sc_e1, sh_e1, CC);
  prep_affine<<<1, 128, 0, stream>>>(ext2_b, ext2_bn, CC, sc_e2, sh_e2, CC);
  prep_affine<<<1, 128, 0, stream>>>(pred1_b, pred1_bn, CC, sc_p1, sh_p1, CC);
  prep_affine<<<1, 16, 0, stream>>>(pred2_b, pred2_bn, 13, sc_p2, sh_p2, 16);

  // ---- 3-NN interpolation -> X0 ----
  interp_kernel<<<dim3(NN / 256, BB), 256, 0, stream>>>(xyz1, xyz2, points1, points2,
                                                        last_pred, X0);

  float* out0 = (float*)d_out;                       // new_points_out [B,128,N]
  float* out1 = out0 + (size_t)BB * CC * NN;         // feat^T         [B,N,128]
  float* out2 = out1 + (size_t)BB * NN * CC;         // pred^T         [B,N,13]

  // ---- fuse: 128 x 397(416) GEMM ----
  gemm_bn_relu<KP0, 8, 1, false, 0, true>
      <<<dim3(NN / 64, BB), 256, 0, stream>>>(X0, Wf, sc_f, sh_f, nullptr, X1, nullptr, CC);
  // ---- ext1 ----
  gemm_bn_relu<CC, 8, 1, false, 0, true>
      <<<dim3(NN / 64, BB), 256, 0, stream>>>(X1, We1, sc_e1, sh_e1, nullptr, H, nullptr, CC);
  // ---- ext2 + residual(X1) -> XR, also emits out0 [B,C,N] ----
  gemm_bn_relu<CC, 8, 1, true, 1, true>
      <<<dim3(NN / 64, BB), 256, 0, stream>>>(H, We2, sc_e2, sh_e2, X1, XR, out0, CC);
  // ---- pred1 -> F, also emits out1 [B,N,128] ----
  gemm_bn_relu<CC, 8, 1, false, 2, true>
      <<<dim3(NN / 64, BB), 256, 0, stream>>>(XR, Wp1, sc_p1, sh_p1, nullptr, F, out1, CC);
  // ---- pred2 (13 of 16 rows) -> out2 [B,N,13] ----
  gemm_bn_relu<CC, 1, 2, false, 2, false>
      <<<dim3(NN / 128, BB), 64, 0, stream>>>(F, Wp2, sc_p2, sh_p2, nullptr, nullptr, out2, 13);
}
